// EdgewiseReduce_90108413870656
// MI455X (gfx1250) — matrix operand order repaired and checked by reading, not compiled, not run on hardware
//
#include <hip/hip_runtime.h>
#include <hip/hip_bf16.h>
#include <cstdint>

#define NN 50000
#define NE 400000
#define EDGE_DIM 144
#define HEADS 16
#define OUTD 512
#define HID 64
#define ISQRTD 0.17677669529663687f

typedef _Float16 v16h __attribute__((ext_vector_type(16)));
typedef float    v8f  __attribute__((ext_vector_type(8)));

__device__ __forceinline__ v8f wmma_f16(v16h a, v16h b, v8f c) {
  return __builtin_amdgcn_wmma_f32_16x16x32_f16(false, a, false, b, (short)0, c, false, false);
}

// ---- WMMA operand layouts (CDNA5 wave32, 16x16x32 f16) ----
// A (16x32, MxK): lane L holds row M=L&15; lanes 0-15: a[0..7]=K0..7, a[8..15]=K16..23;
//                 lanes 16-31: a[0..7]=K8..15, a[8..15]=K24..31.
// B (32x16, KxN): lane L holds col N=L&15; b[0..15] use the same k order as A (K0..7,16..23 / K8..15,24..31).
// C/D (16x16 f32): VGPR r, lanes 0-15 -> M=r, lanes 16-31 -> M=r+8; N = lane&15.

__device__ __forceinline__ v16h load_A_gf32(const float* __restrict__ X, int ld,
                                            int row0, int kbase, int lane) {
  int row  = row0 + (lane & 15);
  int koff = (lane >> 4) << 3;
  const float* p = X + (size_t)row * ld + kbase + koff;
  v16h a;
#pragma unroll
  for (int i = 0; i < 8; ++i) { a[i] = (_Float16)p[i]; a[8 + i] = (_Float16)p[16 + i]; }
  return a;
}

__device__ __forceinline__ v16h load_A_lds(const _Float16* H, int ld, int kbase, int lane) {
  int row  = lane & 15;
  int koff = (lane >> 4) << 3;
  const _Float16* p = H + row * ld + kbase + koff;
  v16h a;
#pragma unroll
  for (int i = 0; i < 8; ++i) { a[i] = p[i]; a[8 + i] = p[16 + i]; }
  return a;
}

// Packed-B: weights pre-swizzled to [tile][kstep][lane][16 halves] -> one 32B vector
// load per lane, coalesced 1KB/wave per tile, zero conversion ALU in the hot loop.
__device__ __forceinline__ v16h load_Bp(const _Float16* __restrict__ base, int t, int s,
                                        int ksteps, int lane) {
  return *(const v16h*)(base + (((size_t)(t * ksteps + s) * 32 + lane) << 4));
}

__device__ __forceinline__ float silu(float x) { return x / (1.f + __expf(-x)); }

__device__ __forceinline__ void atomicMaxF32(float* addr, float v) {
  if (v >= 0.f) atomicMax((int*)addr, __float_as_int(v));
  else          atomicMin((unsigned int*)addr, __float_as_uint(v));
}

// ---------------- weight pre-pack (f32 row-major -> f16 WMMA-B layout) ----------------
__device__ __forceinline__ void pack_one(const float* __restrict__ W, int O, int Iactual,
                                         int ksteps, _Float16* __restrict__ dst, int p) {
  int per_tile = ksteps * 32 * 16;
  int t    = p / per_tile;
  int rem  = p - t * per_tile;
  int s    = rem >> 9;            // / (32*16)
  int rem2 = rem & 511;
  int l    = rem2 >> 4;
  int i    = rem2 & 15;
  int k    = s * 32 + ((i >> 3) << 4) + ((l >> 4) << 3) + (i & 7);
  int col  = t * 16 + (l & 15);
  float v  = (k < Iactual) ? W[(size_t)k * O + col] : 0.f;   // zero-pad (kw1: K 16->32)
  dst[p] = (_Float16)v;
}

// packed-half offsets: qw1 0 | qw2 4096 | qw3 8192 | kw1 40960 | kw2 43008 | kw3 47104
#define PK_TOTAL 79872
__global__ void pack_weights_kernel(const float* __restrict__ qw1, const float* __restrict__ qw2,
                                    const float* __restrict__ qw3, const float* __restrict__ kw1,
                                    const float* __restrict__ kw2, const float* __restrict__ kw3,
                                    _Float16* __restrict__ pw) {
  int idx = blockIdx.x * blockDim.x + threadIdx.x;
  if (idx >= PK_TOTAL) return;
  if (idx < 4096)       { pack_one(qw1, 64,  64, 2, pw,          idx);          return; }
  if (idx < 8192)       { pack_one(qw2, 64,  64, 2, pw + 4096,   idx - 4096);   return; }
  if (idx < 40960)      { pack_one(qw3, 512, 64, 2, pw + 8192,   idx - 8192);   return; }
  if (idx < 43008)      { pack_one(kw1, 64,  16, 1, pw + 40960,  idx - 40960);  return; }
  if (idx < 47104)      { pack_one(kw2, 64,  64, 2, pw + 43008,  idx - 43008);  return; }
                          pack_one(kw3, 512, 64, 2, pw + 47104,  idx - 47104);
}

// ---------------- init ----------------
__global__ void init_kernel(float* __restrict__ out, float* __restrict__ segmax,
                            float* __restrict__ denom) {
  int idx = blockIdx.x * blockDim.x + threadIdx.x;
  if (idx < NN * EDGE_DIM) out[idx] = 0.f;
  if (idx < NN * HEADS) { segmax[idx] = -__builtin_inff(); denom[idx] = 0.f; }
}

// ---------------- Q MLP per node tile (16 rows / wave) ----------------
__global__ void __launch_bounds__(32) qmlp_kernel(
    const float* __restrict__ X,
    const float* __restrict__ b1, const float* __restrict__ b2, const float* __restrict__ b3,
    const _Float16* __restrict__ pw1, const _Float16* __restrict__ pw2,
    const _Float16* __restrict__ pw3, _Float16* __restrict__ Qout) {
  __shared__ _Float16 Ha[16][HID];
  __shared__ _Float16 Hb[16][HID];
  const int lane  = threadIdx.x;
  const int node0 = blockIdx.x * 16;
  const int rbase = (lane >> 4) * 8;
  const int n     = lane & 15;

  // warm L0/L2 for the big layer-3 weight table (global_prefetch_b8)
  __builtin_prefetch((const char*)pw3 + lane * 2048, 0, 0);

  // layer 1: 64 -> 64
  v16h a0 = load_A_gf32(X, HID, node0, 0, lane);
  v16h a1 = load_A_gf32(X, HID, node0, 32, lane);
#pragma unroll
  for (int t = 0; t < 4; ++t) {
    v8f c = {0, 0, 0, 0, 0, 0, 0, 0};
    c = wmma_f16(a0, load_Bp(pw1, t, 0, 2, lane), c);
    c = wmma_f16(a1, load_Bp(pw1, t, 1, 2, lane), c);
    int col = t * 16 + n; float bv = b1[col];
#pragma unroll
    for (int r = 0; r < 8; ++r) Ha[rbase + r][col] = (_Float16)silu(c[r] + bv);
  }
  __syncthreads();

  // layer 2: 64 -> 64
  a0 = load_A_lds(&Ha[0][0], HID, 0, lane);
  a1 = load_A_lds(&Ha[0][0], HID, 32, lane);
#pragma unroll
  for (int t = 0; t < 4; ++t) {
    v8f c = {0, 0, 0, 0, 0, 0, 0, 0};
    c = wmma_f16(a0, load_Bp(pw2, t, 0, 2, lane), c);
    c = wmma_f16(a1, load_Bp(pw2, t, 1, 2, lane), c);
    int col = t * 16 + n; float bv = b2[col];
#pragma unroll
    for (int r = 0; r < 8; ++r) Hb[rbase + r][col] = (_Float16)silu(c[r] + bv);
  }
  __syncthreads();

  // layer 3: 64 -> 512, Q written f16 (51.2MB table, L2-resident for edge gathers)
  a0 = load_A_lds(&Hb[0][0], HID, 0, lane);
  a1 = load_A_lds(&Hb[0][0], HID, 32, lane);
  for (int t = 0; t < 32; ++t) {
    v8f c = {0, 0, 0, 0, 0, 0, 0, 0};
    c = wmma_f16(a0, load_Bp(pw3, t, 0, 2, lane), c);
    c = wmma_f16(a1, load_Bp(pw3, t, 1, 2, lane), c);
    int col = t * 16 + n; float bv = b3[col];
#pragma unroll
    for (int r = 0; r < 8; ++r)
      Qout[(size_t)(node0 + rbase + r) * OUTD + col] = (_Float16)(c[r] + bv);
  }
}

// ---------------- K MLP per edge tile + fused Q.K dot + segment max ----------------
__global__ void __launch_bounds__(32) kmlp_kernel(
    const int* __restrict__ ei, const float* __restrict__ EF,
    const float* __restrict__ b1, const float* __restrict__ b2, const float* __restrict__ b3,
    const _Float16* __restrict__ pw1, const _Float16* __restrict__ pw2,
    const _Float16* __restrict__ pw3, const _Float16* __restrict__ Qf,
    float* __restrict__ Wbuf, float* __restrict__ segmax) {
  __shared__ _Float16 Ha[16][HID];
  __shared__ _Float16 Hb[16][HID];
  const int lane  = threadIdx.x;
  const int e0    = blockIdx.x * 16;
  const int rbase = (lane >> 4) * 8;
  const int n     = lane & 15;

  // warm L0/L2 for the big layer-3 weight table (global_prefetch_b8)
  __builtin_prefetch((const char*)pw3 + lane * 2048, 0, 0);

  int ctr[8];
#pragma unroll
  for (int r = 0; r < 8; ++r) ctr[r] = ei[e0 + rbase + r];  // edge_index[0][e]

  // layer 1: in=16 padded to 32 (A upper half zero; packed B already zero-padded)
  v16h a0;
  {
    int row  = e0 + n;
    int koff = (lane >> 4) * 8;
#pragma unroll
    for (int i = 0; i < 8; ++i) {
      a0[i]     = (_Float16)EF[(size_t)row * EDGE_DIM + koff + i];
      a0[8 + i] = (_Float16)0.f;
    }
  }
#pragma unroll
  for (int t = 0; t < 4; ++t) {
    v8f c = {0, 0, 0, 0, 0, 0, 0, 0};
    c = wmma_f16(a0, load_Bp(pw1, t, 0, 1, lane), c);
    int col = t * 16 + n; float bv = b1[col];
#pragma unroll
    for (int r = 0; r < 8; ++r) Ha[rbase + r][col] = (_Float16)silu(c[r] + bv);
  }
  __syncthreads();

  // layer 2: 64 -> 64
  v16h h0 = load_A_lds(&Ha[0][0], HID, 0, lane);
  v16h h1 = load_A_lds(&Ha[0][0], HID, 32, lane);
#pragma unroll
  for (int t = 0; t < 4; ++t) {
    v8f c = {0, 0, 0, 0, 0, 0, 0, 0};
    c = wmma_f16(h0, load_Bp(pw2, t, 0, 2, lane), c);
    c = wmma_f16(h1, load_Bp(pw2, t, 1, 2, lane), c);
    int col = t * 16 + n; float bv = b2[col];
#pragma unroll
    for (int r = 0; r < 8; ++r) Hb[rbase + r][col] = (_Float16)silu(c[r] + bv);
  }
  __syncthreads();

  // layer 3: 64 -> 512, fused per-head dot with gathered Q (f16, L2-resident).
  // Per-lane partials for both 16-col tiles of a head are summed BEFORE the
  // cross-lane reduction -> one 4-step shfl_xor tree per head (not per tile).
  h0 = load_A_lds(&Hb[0][0], HID, 0, lane);
  h1 = load_A_lds(&Hb[0][0], HID, 32, lane);
  float pp[8];
  for (int t = 0; t < 32; ++t) {
    v8f c = {0, 0, 0, 0, 0, 0, 0, 0};
    c = wmma_f16(h0, load_Bp(pw3, t, 0, 2, lane), c);
    c = wmma_f16(h1, load_Bp(pw3, t, 1, 2, lane), c);
    int col = t * 16 + n; float bv = b3[col];
    if ((t & 1) == 0) {
#pragma unroll
      for (int r = 0; r < 8; ++r) {
        float kv = c[r] + bv;
        float qv = (float)Qf[(size_t)ctr[r] * OUTD + col];
        pp[r] = kv * qv;
      }
    } else {
#pragma unroll
      for (int r = 0; r < 8; ++r) {
        float kv = c[r] + bv;
        float qv = (float)Qf[(size_t)ctr[r] * OUTD + col];
        pp[r] += kv * qv;
      }
      // reduce across the 16-lane half (rows are constant within a half)
#pragma unroll
      for (int off = 1; off < 16; off <<= 1) {
#pragma unroll
        for (int r = 0; r < 8; ++r) pp[r] += __shfl_xor(pp[r], off, 32);
      }
      int head = t >> 1;
      if (n == 0) {
#pragma unroll
        for (int r = 0; r < 8; ++r) {
          float wv = pp[r] * ISQRTD;
          int e = e0 + rbase + r;
          Wbuf[(size_t)e * HEADS + head] = wv;
          atomicMaxF32(&segmax[(size_t)ctr[r] * HEADS + head], wv);
        }
      }
    }
  }
}

// ---------------- exp + denominators ----------------
__global__ void exp_kernel(const int* __restrict__ ei, const float* __restrict__ Win,
                           const float* __restrict__ segmax, float* __restrict__ ex,
                           float* __restrict__ denom) {
  int idx = blockIdx.x * blockDim.x + threadIdx.x;
  if (idx >= NE * HEADS) return;
  int e = idx >> 4, h = idx & 15;
  int c = ei[e];
  float x = __expf(Win[idx] - segmax[c * HEADS + h]);
  ex[idx] = x;
  atomicAdd(&denom[c * HEADS + h], x);
}

// ---------------- weighted scatter-add into out[node,144] ----------------
__global__ void scatter_kernel(const int* __restrict__ ei, const float* __restrict__ EF,
                               const float* __restrict__ ex, const float* __restrict__ denom,
                               float* __restrict__ out) {
  int idx = blockIdx.x * blockDim.x + threadIdx.x;
  if (idx >= NE * EDGE_DIM) return;
  int e = idx / EDGE_DIM;
  int j = idx - e * EDGE_DIM;
  int c = ei[e];
  int h = (j < 16) ? j : ((j < 64) ? (j - 16) / 3 : (j - 64) / 5);
  float attn = ex[e * HEADS + h] / denom[c * HEADS + h];
  atomicAdd(&out[(size_t)c * EDGE_DIM + j], EF[idx] * attn);
}

extern "C" void kernel_launch(void* const* d_in, const int* in_sizes, int n_in,
                              void* d_out, int out_size, void* d_ws, size_t ws_size,
                              hipStream_t stream) {
  const int*   ei  = (const int*)d_in[0];
  const float* EF  = (const float*)d_in[1];
  const float* NA  = (const float*)d_in[2];
  const float* qw1 = (const float*)d_in[3];  const float* qb1 = (const float*)d_in[4];
  const float* qw2 = (const float*)d_in[5];  const float* qb2 = (const float*)d_in[6];
  const float* qw3 = (const float*)d_in[7];  const float* qb3 = (const float*)d_in[8];
  const float* kw1 = (const float*)d_in[9];  const float* kb1 = (const float*)d_in[10];
  const float* kw2 = (const float*)d_in[11]; const float* kb2 = (const float*)d_in[12];
  const float* kw3 = (const float*)d_in[13]; const float* kb3 = (const float*)d_in[14];
  float* out = (float*)d_out;

  // workspace layout (~83.36 MB): Qf16 | W/ex | segmax | denom | packed weights
  char* ws = (char*)d_ws;
  _Float16* Qf    = (_Float16*)ws;              // 50000*512*2 = 51,200,000 B
  float* Wbuf     = (float*)(ws + 51200000);    // 400000*16*4 = 25,600,000 B
  float* segmax   = (float*)(ws + 76800000);    // 50000*16*4  =  3,200,000 B
  float* denom    = (float*)(ws + 80000000);    // 50000*16*4  =  3,200,000 B
  _Float16* pw    = (_Float16*)(ws + 83200000); // 79872*2     =    159,744 B (32B aligned)
  _Float16* pqw1  = pw;
  _Float16* pqw2  = pw + 4096;
  _Float16* pqw3  = pw + 8192;
  _Float16* pkw1  = pw + 40960;
  _Float16* pkw2  = pw + 43008;
  _Float16* pkw3  = pw + 47104;

  pack_weights_kernel<<<(PK_TOTAL + 255) / 256, 256, 0, stream>>>(qw1, qw2, qw3, kw1, kw2, kw3, pw);
  init_kernel<<<(NN * EDGE_DIM + 255) / 256, 256, 0, stream>>>(out, segmax, denom);
  qmlp_kernel<<<NN / 16, 32, 0, stream>>>(NA, qb1, qb2, qb3, pqw1, pqw2, pqw3, Qf);
  kmlp_kernel<<<NE / 16, 32, 0, stream>>>(ei, EF, kb1, kb2, kb3, pkw1, pkw2, pkw3, Qf, Wbuf, segmax);
  exp_kernel<<<(NE * HEADS + 255) / 256, 256, 0, stream>>>(ei, Wbuf, segmax, Wbuf, denom);
  scatter_kernel<<<(NE * EDGE_DIM + 255) / 256, 256, 0, stream>>>(ei, EF, Wbuf, denom, out);
}
